// LinkCouplingFlow_63591285784761
// MI455X (gfx1250) — matrix operand order repaired and checked by reading, not compiled, not run on hardware
//
#include <hip/hip_runtime.h>

// ---------- types ----------
typedef __bf16 bf16_t;
typedef __attribute__((ext_vector_type(8)))  bf16_t v8bf;
typedef __attribute__((ext_vector_type(16))) bf16_t v16bf;
typedef __attribute__((ext_vector_type(8)))  float  v8f;

#define TWO_PI_F   6.283185307179586f
#define INV_2PI_F  0.15915494309189535f
#define L_LEN      3072
#define N_SITE     1024
#define H_DIM      256
#define K_MIX      16
#define P_DIM      48

#define WAVES      4                    // waves per block
#define TPW        2                    // 16-row tiles per wave
#define TILES      (WAVES * TPW)        // 8 tiles -> 128 rows per block
#define BLOCK_ROWS (TILES * 16)         // 128
#define NBLOCKS    (65536 / BLOCK_ROWS) // 512 blocks per transform

// ---- LDS carve (single char arena; f32 params overlay the dead A region) ----
#define OFF_A      0u                         // bf16 [128][256]  = 65536 B
#define OFF_H      65536u                     // bf16 [128][256]  = 65536 B
#define OFF_B      131072u                    // bf16 [16][256]   =  8192 B  (staged weight panel)
#define OFF_C      139264u                    // f32  [4][128]    =  2048 B  (c1,c2,u,zr)
#define SMEM_BYTES 141312u

__device__ __forceinline__ float mod2pi(float x) {
    float y = x - floorf(x * INV_2PI_F) * TWO_PI_F;
    if (y >= TWO_PI_F) y -= TWO_PI_F;
    if (y < 0.0f)      y += TWO_PI_F;
    return y;
}

__device__ __forceinline__ v16bf join16(v8bf lo, v8bf hi) {
    return __builtin_shufflevector(lo, hi, 0,1,2,3,4,5,6,7,8,9,10,11,12,13,14,15);
}

// Stage an 8KB (16 x 256 bf16) weight panel into LDS with CDNA5 async copy:
// GLOBAL_LOAD_ASYNC_TO_LDS_B128 (ASYNCcnt-tracked), then drain ASYNCcnt.
__device__ __forceinline__ void stage_panel_async(const bf16_t* __restrict__ gsrc,
                                                  unsigned char* smem, int tid) {
    const char* gb = (const char*)gsrc;
#pragma unroll
    for (int it = 0; it < 4; ++it) {
        unsigned byte_off = (unsigned)(it * 128 + tid) * 16u;
        unsigned lds_addr = (unsigned)(uintptr_t)(smem + OFF_B + byte_off);
        const void* gp    = gb + byte_off;
        asm volatile("global_load_async_to_lds_b128 %0, %1, off"
                     :: "v"(lds_addr), "v"(gp) : "memory");
    }
    asm volatile("s_wait_asynccnt 0" ::: "memory");
}

// ---------- prep: transpose+bf16 weights, init z/ldj ----------
__global__ __launch_bounds__(256)
void flow_prep_kernel(const float* __restrict__ z_in,
                      const float* __restrict__ W2,
                      const float* __restrict__ W3,
                      float* __restrict__ z_out,
                      float* __restrict__ ldj,
                      bf16_t* __restrict__ W2T,
                      bf16_t* __restrict__ W3T)
{
    int gid = blockIdx.x * blockDim.x + threadIdx.x;
    if (gid < 12 * H_DIM * H_DIM) {            // W2T[t][n][k] <- W2[t][k][n]
        int t = gid / (H_DIM * H_DIM);
        int r = gid % (H_DIM * H_DIM);
        int n = r / H_DIM, k = r % H_DIM;
        W2T[gid] = (bf16_t)W2[t * H_DIM * H_DIM + k * H_DIM + n];
    }
    if (gid < 12 * P_DIM * H_DIM) {            // W3T[t][n][k] <- W3[t][k][n]
        int t = gid / (P_DIM * H_DIM);
        int r = gid % (P_DIM * H_DIM);
        int n = r / H_DIM, k = r % H_DIM;
        W3T[gid] = (bf16_t)W3[t * H_DIM * P_DIM + k * P_DIM + n];
    }
    if (gid < 64 * L_LEN) z_out[gid] = z_in[gid];
    if (gid < 64)         ldj[gid]   = 0.0f;
}

// ---------- one coupling transform ----------
__global__ __launch_bounds__(BLOCK_ROWS)
void flow_step_kernel(float* __restrict__ z, float* __restrict__ ldj,
                      const float* __restrict__ W1, const float* __restrict__ b1,
                      const float* __restrict__ b2, const float* __restrict__ b3,
                      const bf16_t* __restrict__ W2T, const bf16_t* __restrict__ W3T,
                      int t, int off)
{
    __shared__ __align__(16) unsigned char smem[SMEM_BYTES];
    bf16_t* sA   = (bf16_t*)(smem + OFF_A);   // [128][256] layer-1 acts
    bf16_t* sH   = (bf16_t*)(smem + OFF_H);   // [128][256] layer-2 acts
    bf16_t* sB   = (bf16_t*)(smem + OFF_B);   // [16][256] staged weight panel
    float*  sCtx = (float*) (smem + OFF_C);   // [4][128]  c1,c2,u,zr
    float*  sP   = (float*) (smem + OFF_A);   // [128][48] params (overlays dead sA)

    const int tid  = threadIdx.x;
    const int lane = tid & 31;
    const int wv   = tid >> 5;
    const int row0 = blockIdx.x * BLOCK_ROWS;
    const int bidx = row0 >> 10;               // 1024 rows per batch, block-aligned
    float* zb = z + bidx * L_LEN;

    const float*  W1t = W1  + t * 2 * H_DIM;
    const float*  b1t = b1  + t * H_DIM;
    const float*  b2t = b2  + t * H_DIM;
    const float*  b3t = b3  + t * P_DIM;
    const bf16_t* W2t = W2T + t * H_DIM * H_DIM;
    const bf16_t* W3t = W3T + t * P_DIM * H_DIM;

    // ---- phase 1: per-row context (one thread per row) ----
    {
        int g   = row0 + tid;
        int i   = g & (N_SITE - 1);
        int idx = 3 * i + off;
        auto zp = [&](int p) -> float {
            p %= L_LEN; if (p < 0) p += L_LEN; return zb[p];
        };
        auto Vp = [&](int p) -> float { return mod2pi(zp(p) - zp(p - 1)); };
        sCtx[0 * BLOCK_ROWS + tid] = Vp(idx - 1);
        sCtx[1 * BLOCK_ROWS + tid] = Vp(idx + 2);
        sCtx[2 * BLOCK_ROWS + tid] = Vp(idx);
        sCtx[3 * BLOCK_ROWS + tid] = zp(idx - 1);
    }
    __syncthreads();

    // ---- phase 2: layer 1 (K=2, scalar) -> sA bf16 ----
    for (int e = tid; e < BLOCK_ROWS * H_DIM; e += BLOCK_ROWS) {
        int r = e >> 8, c = e & 255;
        float v = sCtx[r] * W1t[c] + sCtx[BLOCK_ROWS + r] * W1t[H_DIM + c] + b1t[c];
        sA[r * H_DIM + c] = (bf16_t)tanhf(v);
    }
    __syncthreads();

    // fragment index helpers (ISA 16-bit A/B layouts, wave32)
    const int arow = lane & 15;
    const int asel = (lane & 16) ? 8  : 0;
    const int bsel = (lane & 16) ? 8  : 0;
    const int ncol = lane & 15;
    const int msel = (lane & 16) ? 8  : 0;

    // ---- phase 3: layer 2, 256x256 GEMM; async-staged B panel, 2 row-tiles/wave ----
    for (int nt = 0; nt < H_DIM / 16; ++nt) {
        stage_panel_async(W2t + nt * 16 * H_DIM, smem, tid);
        __syncthreads();
        v8f acc[TPW] = {};
#pragma unroll
        for (int kc = 0; kc < H_DIM / 32; ++kc) {
            const bf16_t* bp = sB + ncol * H_DIM + kc * 32 + bsel;
            v16bf bm = join16(*(const v8bf*)bp, *(const v8bf*)(bp + 16));
#pragma unroll
            for (int j = 0; j < TPW; ++j) {
                int rg = (wv * TPW + j) * 16 + arow;
                const bf16_t* ap = sA + rg * H_DIM + kc * 32 + asel;
                v16bf a = join16(*(const v8bf*)ap, *(const v8bf*)(ap + 16));
                acc[j] = __builtin_amdgcn_wmma_f32_16x16x32_bf16(
                             false, a, false, bm, (short)0, acc[j], false, false);
            }
        }
        int cn = nt * 16 + ncol;
        float bias = b2t[cn];
#pragma unroll
        for (int j = 0; j < TPW; ++j)
#pragma unroll
            for (int v = 0; v < 8; ++v) {
                int m = (wv * TPW + j) * 16 + v + msel;
                sH[m * H_DIM + cn] = (bf16_t)tanhf(acc[j][v] + bias);
            }
        __syncthreads();   // protect sB before next stage overwrites it
    }

    // ---- phase 4: layer 3, 256x48 GEMM -> sP (overlaid on sA) ----
    for (int nt = 0; nt < P_DIM / 16; ++nt) {
        stage_panel_async(W3t + nt * 16 * H_DIM, smem, tid);
        __syncthreads();
        v8f acc[TPW] = {};
#pragma unroll
        for (int kc = 0; kc < H_DIM / 32; ++kc) {
            const bf16_t* bp = sB + ncol * H_DIM + kc * 32 + bsel;
            v16bf bm = join16(*(const v8bf*)bp, *(const v8bf*)(bp + 16));
#pragma unroll
            for (int j = 0; j < TPW; ++j) {
                int rg = (wv * TPW + j) * 16 + arow;
                const bf16_t* ap = sH + rg * H_DIM + kc * 32 + asel;
                v16bf a = join16(*(const v8bf*)ap, *(const v8bf*)(ap + 16));
                acc[j] = __builtin_amdgcn_wmma_f32_16x16x32_bf16(
                             false, a, false, bm, (short)0, acc[j], false, false);
            }
        }
        int cn = nt * 16 + ncol;
        float bias = b3t[cn];
#pragma unroll
        for (int j = 0; j < TPW; ++j)
#pragma unroll
            for (int v = 0; v < 8; ++v) {
                int m = (wv * TPW + j) * 16 + v + msel;
                sP[m * P_DIM + cn] = acc[j][v] + bias;
            }
        __syncthreads();
    }

    // ---- phase 5: mixture-of-sigmoids transform + LDJ (one thread per row) ----
    {
        int g   = row0 + tid;
        int i   = g & (N_SITE - 1);
        int idx = 3 * i + off;
        const float* pr = sP + tid * P_DIM;

        float u = sCtx[2 * BLOCK_ROWS + tid] * INV_2PI_F;
        u = fminf(fmaxf(u, 1e-6f), 1.0f - 1e-6f);
        float a2  = u * u;
        float b2_ = (1.0f - u) * (1.0f - u);
        float apb = a2 + b2_;
        float s   = a2 / apb;
        float ds_du = 2.0f * u * (1.0f - u) / (apb * apb);
        s = fminf(fmaxf(s, 1e-6f), 1.0f - 1e-6f);
        float logit_s = logf(s) - log1pf(-s);

        float wmax = -3.0e38f;
#pragma unroll
        for (int k = 0; k < K_MIX; ++k) wmax = fmaxf(wmax, pr[32 + k]);
        float we[K_MIX]; float wsum = 0.0f;
#pragma unroll
        for (int k = 0; k < K_MIX; ++k) { we[k] = expf(pr[32 + k] - wmax); wsum += we[k]; }
        float inv_wsum = 1.0f / wsum;

        float y = 0.0f, dyds = 0.0f;
#pragma unroll
        for (int k = 0; k < K_MIX; ++k) {
            float wgt   = we[k] * inv_wsum * (1.0f - K_MIX * 0.01f) + 0.01f;
            float alpha = expf(pr[k]);
            float shift = pr[16 + k];
            float gg    = 1.0f / (1.0f + expf(-alpha * (logit_s + shift)));
            y    += wgt * gg;
            dyds += wgt * alpha * gg * (1.0f - gg);
        }
        dyds /= (s * (1.0f - s));
        float ldj_r = logf(dyds * ds_du);
        zb[idx] = mod2pi(TWO_PI_F * y + sCtx[3 * BLOCK_ROWS + tid]);
        atomicAdd(&ldj[bidx], ldj_r);
    }
}

// ---------- host launcher ----------
extern "C" void kernel_launch(void* const* d_in, const int* in_sizes, int n_in,
                              void* d_out, int out_size, void* d_ws, size_t ws_size,
                              hipStream_t stream) {
    const float* z_in = (const float*)d_in[0];
    const float* W1   = (const float*)d_in[1];
    const float* b1   = (const float*)d_in[2];
    const float* W2   = (const float*)d_in[3];
    const float* b2   = (const float*)d_in[4];
    const float* W3   = (const float*)d_in[5];
    const float* b3   = (const float*)d_in[6];

    float* z_out = (float*)d_out;                 // working z lives in d_out
    float* ldj   = z_out + 64 * L_LEN;

    bf16_t* W2T = (bf16_t*)d_ws;                                       // 12*256*256 bf16
    bf16_t* W3T = (bf16_t*)((char*)d_ws + (size_t)12 * 256 * 256 * 2); // 12*48*256 bf16

    flow_prep_kernel<<<(12 * 256 * 256 + 255) / 256, 256, 0, stream>>>(
        z_in, W2, W3, z_out, ldj, W2T, W3T);

    for (int t = 0; t < 12; ++t) {
        int m   = t % 3;
        int off = (m == 0) ? 0 : ((m == 1) ? 2 : 1);
        flow_step_kernel<<<NBLOCKS, BLOCK_ROWS, 0, stream>>>(
            z_out, ldj, W1, b1, b2, b3, W2T, W3T, t, off);
    }
}
